// Decoder_7911329759944
// MI455X (gfx1250) — compile-verified, matrix operand order checked by reading, not run on hardware
//
#include <hip/hip_runtime.h>
#include <hip/hip_fp16.h>

// ---------------- problem constants ----------------
#define NWG   64
#define TPB   256
#define NT    (NWG*TPB)

#define B_    64
#define TENC  400
#define TDEC  500
#define NMEL  80
#define EDIM  512
#define ADIM  128
#define PREN  256
#define KA    1792      // PREN + EDIM + 1024  (attention LSTM fused K)
#define KD    2560      // 1024 + EDIM + 1024  (decoder  LSTM fused K)
#define NGATE 4096

typedef _Float16     v16h  __attribute__((ext_vector_type(16)));
typedef float        v8f   __attribute__((ext_vector_type(8)));
typedef unsigned int u32x4 __attribute__((ext_vector_type(4)));
typedef int          i32x8 __attribute__((ext_vector_type(8)));
typedef int          i32x4 __attribute__((ext_vector_type(4)));

// ---------------- workspace layout (bytes) ----------------
constexpr size_t WS_BAR = 0;                                   // barrier: gen, cnt
constexpr size_t WS_WA  = 256;                                 // f16 [4096][1792]
constexpr size_t WS_WD  = WS_WA  + (size_t)NGATE*KA*2;         // f16 [4096][2560]
constexpr size_t WS_BA  = WS_WD  + (size_t)NGATE*KD*2;         // f32 [4096] (bih+bhh)
constexpr size_t WS_BD  = WS_BA  + (size_t)NGATE*4;
constexpr size_t WS_PM  = WS_BD  + (size_t)NGATE*4;            // f32 [64][400][128]
constexpr size_t WS_XA  = WS_PM  + (size_t)B_*TENC*ADIM*4;     // f16 [64][1792]
constexpr size_t WS_XD  = WS_XA  + (size_t)B_*KA*2;            // f16 [64][2560]
constexpr size_t WS_P1  = WS_XD  + (size_t)B_*KD*2;            // f32 [64][256]
constexpr size_t WS_AH  = WS_P1  + (size_t)B_*PREN*4;          // f32 [64][1024]
constexpr size_t WS_AC  = WS_AH  + (size_t)B_*1024*4;
constexpr size_t WS_DH  = WS_AC  + (size_t)B_*1024*4;
constexpr size_t WS_DC  = WS_DH  + (size_t)B_*1024*4;
constexpr size_t WS_AW  = WS_DC  + (size_t)B_*1024*4;          // f32 [64][400]
constexpr size_t WS_AWC = WS_AW  + (size_t)B_*TENC*4;
constexpr size_t WS_CTX = WS_AWC + (size_t)B_*TENC*4;          // f32 [64][512]
constexpr size_t WS_Q   = WS_CTX + (size_t)B_*EDIM*4;          // f32 [64][128]
constexpr size_t WS_EN  = WS_Q   + (size_t)B_*ADIM*4;          // f32 [64][400]

// output layout (floats): mels | gates | aligns
constexpr size_t OUT_MEL   = 0;
constexpr size_t OUT_GATE  = (size_t)B_*TDEC*NMEL;             // 2,560,000
constexpr size_t OUT_ALIGN = OUT_GATE + (size_t)B_*TDEC;       // 2,592,000

// ---------------- small helpers ----------------
static __device__ __forceinline__ float sigm(float x) { return 1.0f / (1.0f + __expf(-x)); }

static __device__ __forceinline__ unsigned rng_hash(unsigned a, unsigned b, unsigned c) {
  unsigned x = a * 0x9E3779B9u ^ b * 0x85EBCA6Bu ^ c * 0xC2B2AE35u;
  x ^= x >> 16; x *= 0x7FEB352Du; x ^= x >> 15; x *= 0x846CA68Bu; x ^= x >> 16;
  return x;
}
static __device__ __forceinline__ float dropf(float v, float p, unsigned t, unsigned slot, unsigned idx) {
  unsigned r = rng_hash(t * 131u + 7u, slot, idx);
  return (r >= (unsigned)(p * 4294967296.0f)) ? v / (1.0f - p) : 0.0f;
}

// ---------------- device-wide sense barrier ----------------
static __device__ __forceinline__ void grid_barrier(unsigned* bar, unsigned* sGen) {
  __syncthreads();
  if (threadIdx.x == 0) {
    __threadfence();
    unsigned g = *sGen;
    if (atomicAdd(&bar[1], 1u) == NWG - 1) {
      __hip_atomic_store(&bar[1], 0u, __ATOMIC_RELAXED, __HIP_MEMORY_SCOPE_AGENT);
      __threadfence();
      __hip_atomic_store(&bar[0], g + 1u, __ATOMIC_RELEASE, __HIP_MEMORY_SCOPE_AGENT);
    } else {
      while (__hip_atomic_load(&bar[0], __ATOMIC_ACQUIRE, __HIP_MEMORY_SCOPE_AGENT) < g + 1u)
        __builtin_amdgcn_s_sleep(1);
    }
    *sGen = g + 1u;
    __threadfence();
  }
  __syncthreads();
}

// ---------------- Tensor Data Mover staging ----------------
#if defined(__has_builtin)
#if __has_builtin(__builtin_amdgcn_tensor_load_to_lds)
#define HAVE_TDM 1
#endif
#endif
#ifndef HAVE_TDM
#define HAVE_TDM 0
#endif

#if HAVE_TDM
// 2-D tile of f16 elements: tileW elems along rows, tileH rows, row stride strideElems.
// 6-arg builtin (clang-23 lane): (u32x4 g0, i32x8 g1, i32x4 g2, i32x4 g3, i32x8 g4, i32 cpol)
static __device__ __forceinline__ void tdm_load_tile_f16(
    unsigned lds_off, const _Float16* gptr,
    unsigned tileW, unsigned tileH, unsigned rowLen, unsigned nRows, unsigned strideElems)
{
  unsigned long long ga = (unsigned long long)(size_t)gptr;
  u32x4 g0;
  g0[0] = 1u;                                              // count=1 valid descriptor
  g0[1] = lds_off;                                         // lds_addr
  g0[2] = (unsigned)ga;                                    // global_addr[31:0]
  g0[3] = (unsigned)((ga >> 32) & 0x1FFFFFFull) | (2u << 30); // addr[56:32] | type=2
  i32x8 g1;
  g1[0] = (int)(1u << 16);                                 // data_size = 1 -> 2 bytes
  g1[1] = (int)((rowLen & 0xFFFFu) << 16);                 // tensor_dim0[15:0]
  g1[2] = (int)((rowLen >> 16) | ((nRows & 0xFFFFu) << 16));
  g1[3] = (int)((nRows >> 16) | (tileW << 16));            // tile_dim0
  g1[4] = (int)(tileH & 0xFFFFu);                          // tile_dim1 (tile_dim2=0)
  g1[5] = (int)strideElems;                                // tensor_dim0_stride[31:0]
  g1[6] = 0; g1[7] = 0;
  i32x4 gz4; gz4[0] = gz4[1] = gz4[2] = gz4[3] = 0;        // groups 2/3: NULL (2-D tensor)
  i32x8 gz8; gz8[0]=gz8[1]=gz8[2]=gz8[3]=gz8[4]=gz8[5]=gz8[6]=gz8[7]=0;
  __builtin_amdgcn_tensor_load_to_lds(g0, g1, gz4, gz4, gz8, 0);
}
#endif

// ---------------- fused LSTM phase (WMMA) ----------------
// gates = [x|h](64xK) @ W^T(Kx4096) ; this WG owns units [wg*16, wg*16+16) of all 4 gates.
static __device__ void lstm_phase(
    const _Float16* __restrict__ X,    // [64][Kdim] fused activations (f16)
    const _Float16* __restrict__ W,    // [4096][Kdim] fused weights (f16, row-major)
    const float*    __restrict__ bias, // [4096] (bih+bhh)
    float* __restrict__ Cst, float* __restrict__ Hst,   // [64][1024]
    _Float16* sink, int sinkLd, int sinkCol,            // optional f16 copy of h
    int Kdim, float pdrop, unsigned t, unsigned slot,
    _Float16* sAf, _Float16* sBf, float* sZf)
{
  const int tid = threadIdx.x;
  const int wg  = blockIdx.x;
  const int lane = tid & 31;
  const int wv   = tid >> 5;
  const int l    = lane & 15;
  const int hi   = lane >> 4;
  const int mtile = wv & 3;      // batch rows mtile*16 .. +16
  const int gp    = wv >> 2;     // 0 -> gates {0,1}, 1 -> gates {2,3}
  const int nch = Kdim >> 5;

  v8f acc0 = {0.f,0.f,0.f,0.f,0.f,0.f,0.f,0.f};
  v8f acc1 = {0.f,0.f,0.f,0.f,0.f,0.f,0.f,0.f};

#if HAVE_TDM
  if (tid == 0) {
    tdm_load_tile_f16((unsigned)(size_t)sAf, X, 32u, 64u, (unsigned)Kdim, 64u, (unsigned)Kdim);
    for (int g = 0; g < 4; ++g)
      tdm_load_tile_f16((unsigned)(size_t)(sBf + g*512),
                        W + (size_t)(g*1024 + wg*16)*Kdim, 32u, 16u,
                        (unsigned)Kdim, 16u, (unsigned)Kdim);
  }
  for (int c = 0; c < nch; ++c) {
    const int buf = c & 1;
    if (tid == 0) __builtin_amdgcn_s_wait_tensorcnt(0);    // chunk c arrived
    __syncthreads();
    if (tid == 0 && (c + 1) < nch) {                       // prefetch chunk c+1
      const int nb = (c + 1) & 1, kc = (c + 1) << 5;
      tdm_load_tile_f16((unsigned)(size_t)(sAf + nb*2048), X + kc,
                        32u, 64u, (unsigned)Kdim, 64u, (unsigned)Kdim);
      for (int g = 0; g < 4; ++g)
        tdm_load_tile_f16((unsigned)(size_t)(sBf + (nb*4 + g)*512),
                          W + (size_t)(g*1024 + wg*16)*Kdim + kc, 32u, 16u,
                          (unsigned)Kdim, 16u, (unsigned)Kdim);
    }
    v16h af, bf0, bf1;
    #pragma unroll
    for (int h = 0; h < 16; ++h) {
      const int j = h >> 1, e = h & 1;
      const int k = 2*j + e + ((j >= 4) ? 8 : 0) + hi*8;   // 16-bit A/B VGPR striping
      af[h]  = sAf[buf*2048 + (mtile*16 + l)*32 + k];
      bf0[h] = sBf[(buf*4 + gp*2 + 0)*512 + l*32 + k];
      bf1[h] = sBf[(buf*4 + gp*2 + 1)*512 + l*32 + k];
    }
    acc0 = __builtin_amdgcn_wmma_f32_16x16x32_f16(false, af, false, bf0, (short)0, acc0, false, false);
    acc1 = __builtin_amdgcn_wmma_f32_16x16x32_f16(false, af, false, bf1, (short)0, acc1, false, false);
    __syncthreads();
  }
#else
  for (int c = 0; c < nch; ++c) {
    const int kc = c << 5;
    __syncthreads();
    for (int i = tid; i < 64*32; i += TPB)
      sAf[i] = X[(size_t)(i >> 5)*Kdim + kc + (i & 31)];
    for (int i = tid; i < 4*16*32; i += TPB) {
      const int g = i >> 9, r = (i >> 5) & 15, kk = i & 31;
      sBf[i] = W[(size_t)(g*1024 + wg*16 + r)*Kdim + kc + kk];
    }
    __syncthreads();
    v16h af, bf0, bf1;
    #pragma unroll
    for (int h = 0; h < 16; ++h) {
      const int j = h >> 1, e = h & 1;
      const int k = 2*j + e + ((j >= 4) ? 8 : 0) + hi*8;
      af[h]  = sAf[(mtile*16 + l)*32 + k];
      bf0[h] = sBf[(gp*2 + 0)*512 + l*32 + k];
      bf1[h] = sBf[(gp*2 + 1)*512 + l*32 + k];
    }
    acc0 = __builtin_amdgcn_wmma_f32_16x16x32_f16(false, af, false, bf0, (short)0, acc0, false, false);
    acc1 = __builtin_amdgcn_wmma_f32_16x16x32_f16(false, af, false, bf1, (short)0, acc1, false, false);
  }
  __syncthreads();
#endif

  // C/D layout: lane l = column N, VGPR r = row M=r (+8 for lanes>=16)
  #pragma unroll
  for (int r = 0; r < 8; ++r) {
    sZf[((gp*2 + 0)*64 + mtile*16 + hi*8 + r)*16 + l] = acc0[r];
    sZf[((gp*2 + 1)*64 + mtile*16 + hi*8 + r)*16 + l] = acc1[r];
  }
  __syncthreads();

  for (int idx = tid; idx < 64*16; idx += TPB) {
    const int b = idx >> 4, ul = idx & 15;
    const int u = wg*16 + ul;
    const float zi = sZf[(0*64 + b)*16 + ul] + bias[u];
    const float zf = sZf[(1*64 + b)*16 + ul] + bias[1024 + u];
    const float zg = sZf[(2*64 + b)*16 + ul] + bias[2048 + u];
    const float zo = sZf[(3*64 + b)*16 + ul] + bias[3072 + u];
    const float c2 = sigm(zf) * Cst[b*1024 + u] + sigm(zi) * tanhf(zg);
    float hh = sigm(zo) * tanhf(c2);
    hh = dropf(hh, pdrop, t, slot, (unsigned)(b*1024 + u));
    Cst[b*1024 + u] = c2;
    Hst[b*1024 + u] = hh;
    if (sink) sink[(size_t)b*sinkLd + sinkCol + u] = (_Float16)hh;
  }
  __syncthreads();
}

// ---------------- init kernel ----------------
__global__ void init_bar_kernel(unsigned* bar) { bar[0] = 0u; bar[1] = 0u; }

// ---------------- persistent decoder kernel ----------------
__global__ void __launch_bounds__(TPB) tacotron_decoder_kernel(
    const float* __restrict__ memory, const float* __restrict__ dec_in,
    const float* __restrict__ pw1, const float* __restrict__ pb1,
    const float* __restrict__ pw2, const float* __restrict__ pb2,
    const float* __restrict__ a_wih, const float* __restrict__ a_whh,
    const float* __restrict__ a_bih, const float* __restrict__ a_bhh,
    const float* __restrict__ qw, const float* __restrict__ mw,
    const float* __restrict__ vw, const float* __restrict__ cw,
    const float* __restrict__ ldw,
    const float* __restrict__ d_wih, const float* __restrict__ d_whh,
    const float* __restrict__ d_bih, const float* __restrict__ d_bhh,
    const float* __restrict__ projw, const float* __restrict__ projb,
    const float* __restrict__ gatew, const float* __restrict__ gateb,
    const int* __restrict__ mlen,
    float* __restrict__ out, char* __restrict__ ws)
{
  __shared__ _Float16 sAf[2*64*32];
  __shared__ _Float16 sBf[2*4*16*32];
  __shared__ float    sZf[4*64*16];
  __shared__ float    sRed[TPB];
  __shared__ unsigned sGen;

  unsigned* bar = (unsigned*)(ws + WS_BAR);
  _Float16* Wa  = (_Float16*)(ws + WS_WA);
  _Float16* Wd  = (_Float16*)(ws + WS_WD);
  float* ba   = (float*)(ws + WS_BA);
  float* bd   = (float*)(ws + WS_BD);
  float* pm   = (float*)(ws + WS_PM);
  _Float16* xa = (_Float16*)(ws + WS_XA);
  _Float16* xd = (_Float16*)(ws + WS_XD);
  float* p1   = (float*)(ws + WS_P1);
  float* ah   = (float*)(ws + WS_AH);
  float* ac   = (float*)(ws + WS_AC);
  float* dh   = (float*)(ws + WS_DH);
  float* dc   = (float*)(ws + WS_DC);
  float* aw   = (float*)(ws + WS_AW);
  float* awc  = (float*)(ws + WS_AWC);
  float* ctx  = (float*)(ws + WS_CTX);
  float* qv   = (float*)(ws + WS_Q);
  float* en   = (float*)(ws + WS_EN);

  const int tid  = threadIdx.x;
  const int gtid = blockIdx.x * TPB + tid;
  if (tid == 0) sGen = 0u;
  __syncthreads();

  // ---------- precompute: zero state, f16 weight fusion, processed_memory ----------
  for (int i = gtid; i < B_*1024; i += NT) { ah[i] = 0.f; ac[i] = 0.f; dh[i] = 0.f; dc[i] = 0.f; }
  for (int i = gtid; i < B_*TENC; i += NT) { aw[i] = 0.f; awc[i] = 0.f; }
  for (int i = gtid; i < B_*EDIM; i += NT) ctx[i] = 0.f;
  for (int i = gtid; i < B_*KA;   i += NT) xa[i] = (_Float16)0.f;
  for (int i = gtid; i < B_*KD;   i += NT) xd[i] = (_Float16)0.f;
  for (size_t i = gtid; i < (size_t)NGATE*KA; i += NT) {
    const int n = (int)(i / KA), k = (int)(i % KA);
    Wa[i] = (_Float16)((k < 768) ? a_wih[(size_t)n*768 + k] : a_whh[(size_t)n*1024 + (k - 768)]);
  }
  for (size_t i = gtid; i < (size_t)NGATE*KD; i += NT) {
    const int n = (int)(i / KD), k = (int)(i % KD);
    Wd[i] = (_Float16)((k < 1536) ? d_wih[(size_t)n*1536 + k] : d_whh[(size_t)n*1024 + (k - 1536)]);
  }
  for (int i = gtid; i < NGATE; i += NT) { ba[i] = a_bih[i] + a_bhh[i]; bd[i] = d_bih[i] + d_bhh[i]; }
  for (int i = gtid; i < B_*TENC*ADIM; i += NT) {           // pm[b][t][a]
    const int a = i % ADIM, tt = (i / ADIM) % TENC, b = i / (ADIM*TENC);
    float s = 0.f;
    const float* mrow = memory + ((size_t)b*TENC + tt)*EDIM;
    for (int e = 0; e < EDIM; ++e) s += mrow[e] * mw[a*EDIM + e];
    pm[i] = s;
  }
  grid_barrier(bar, &sGen);

  // ---------- 500 recurrent steps ----------
  for (unsigned t = 0; t < TDEC; ++t) {
    // P0: prenet layer 1
    for (int i = gtid; i < B_*PREN; i += NT) {
      const int j = i % PREN, b = i / PREN;
      float s = pb1[j];
      const float* x = dec_in + ((size_t)b*TDEC + t)*NMEL;
      for (int k = 0; k < NMEL; ++k) s += x[k] * pw1[j*NMEL + k];
      p1[i] = dropf(fmaxf(s, 0.f), 0.5f, t, 0u, (unsigned)i);
    }
    grid_barrier(bar, &sGen);

    // P1: prenet layer 2 -> xa[:, 0:256]
    for (int i = gtid; i < B_*PREN; i += NT) {
      const int j = i % PREN, b = i / PREN;
      float s = pb2[j];
      for (int k = 0; k < PREN; ++k) s += p1[b*PREN + k] * pw2[j*PREN + k];
      s = dropf(fmaxf(s, 0.f), 0.5f, t, 1u, (unsigned)i);
      xa[(size_t)b*KA + j] = (_Float16)s;
    }
    grid_barrier(bar, &sGen);

    // P2: attention LSTM (WMMA); writes ah/ac + xd[:, 0:1024] = f16(ah)
    lstm_phase(xa, Wa, ba, ac, ah, xd, KD, 0, KA, 0.1f, t, 2u, sAf, sBf, sZf);
    grid_barrier(bar, &sGen);

    // P3: attention query q = ah @ query_w^T, and xa[:, 768:1792] = f16(ah)
    for (int i = gtid; i < B_*ADIM; i += NT) {
      const int a = i % ADIM, b = i / ADIM;
      float s = 0.f;
      for (int k = 0; k < 1024; ++k) s += ah[b*1024 + k] * qw[a*1024 + k];
      qv[i] = s;
    }
    for (int i = gtid; i < B_*1024; i += NT) {
      const int b = i >> 10, u = i & 1023;
      xa[(size_t)b*KA + 768 + u] = (_Float16)ah[i];
    }
    grid_barrier(bar, &sGen);

    // P4: location conv (aw, awc) + tanh energies + mask
    for (int i = gtid; i < B_*TENC; i += NT) {
      const int tp = i % TENC, b = i / TENC;
      float loc[32];
      #pragma unroll
      for (int f = 0; f < 32; ++f) loc[f] = 0.f;
      for (int kk = 0; kk < 31; ++kk) {
        const int pos = tp - 15 + kk;
        if (pos < 0 || pos >= TENC) continue;
        const float a0 = aw[b*TENC + pos], a1 = awc[b*TENC + pos];
        #pragma unroll
        for (int f = 0; f < 32; ++f)
          loc[f] += a0 * cw[f*62 + kk] + a1 * cw[f*62 + 31 + kk];
      }
      float e = 0.f;
      const float* pmr = pm + ((size_t)b*TENC + tp)*ADIM;
      for (int a = 0; a < ADIM; ++a) {
        float pl = 0.f;
        #pragma unroll
        for (int f = 0; f < 32; ++f) pl += ldw[a*32 + f] * loc[f];
        e += vw[a] * tanhf(qv[b*ADIM + a] + pmr[a] + pl);
      }
      if (tp >= mlen[b]) e = -1e9f;
      en[i] = e;
    }
    grid_barrier(bar, &sGen);

    // P5: softmax -> aw, awc; context -> ctx, xa[:,256:768], xd[:,1024:1536] (block b)
    {
      const int b = blockIdx.x;
      float m = -3.0e38f;
      for (int tp = tid; tp < TENC; tp += TPB) m = fmaxf(m, en[b*TENC + tp]);
      sRed[tid] = m; __syncthreads();
      for (int s = TPB/2; s > 0; s >>= 1) { if (tid < s) sRed[tid] = fmaxf(sRed[tid], sRed[tid + s]); __syncthreads(); }
      const float mx = sRed[0]; __syncthreads();
      float ssum = 0.f;
      for (int tp = tid; tp < TENC; tp += TPB) ssum += __expf(en[b*TENC + tp] - mx);
      sRed[tid] = ssum; __syncthreads();
      for (int s = TPB/2; s > 0; s >>= 1) { if (tid < s) sRed[tid] += sRed[tid + s]; __syncthreads(); }
      const float inv = 1.f / sRed[0]; __syncthreads();
      for (int tp = tid; tp < TENC; tp += TPB) {
        const float w = __expf(en[b*TENC + tp] - mx) * inv;
        aw[b*TENC + tp] = w;
        awc[b*TENC + tp] += w;
      }
      __threadfence_block();
      __syncthreads();
      for (int e0 = tid; e0 < EDIM; e0 += TPB) {
        float s = 0.f;
        for (int tt = 0; tt < TENC; ++tt)
          s += aw[b*TENC + tt] * memory[((size_t)b*TENC + tt)*EDIM + e0];
        ctx[b*EDIM + e0] = s;
        xa[(size_t)b*KA + 256 + e0]  = (_Float16)s;
        xd[(size_t)b*KD + 1024 + e0] = (_Float16)s;
      }
    }
    grid_barrier(bar, &sGen);

    // P6: decoder LSTM (WMMA)
    lstm_phase(xd, Wd, bd, dc, dh, nullptr, 0, 0, KD, 0.1f, t, 3u, sAf, sBf, sZf);
    grid_barrier(bar, &sGen);

    // P7: projections + outputs + xd[:,1536:2560] = f16(dh)
    for (int i = gtid; i < B_*NMEL; i += NT) {
      const int m = i % NMEL, b = i / NMEL;
      float s = projb[m];
      for (int k = 0; k < 1024; ++k) s += dh[b*1024 + k] * projw[m*1536 + k];
      for (int k = 0; k < EDIM; ++k) s += ctx[b*EDIM + k] * projw[m*1536 + 1024 + k];
      out[OUT_MEL + ((size_t)b*TDEC + t)*NMEL + m] = s;
    }
    for (int i = gtid; i < B_; i += NT) {
      float s = gateb[0];
      for (int k = 0; k < 1024; ++k) s += dh[i*1024 + k] * gatew[k];
      for (int k = 0; k < EDIM; ++k) s += ctx[i*EDIM + k] * gatew[1024 + k];
      out[OUT_GATE + (size_t)i*TDEC + t] = s;
    }
    for (int i = gtid; i < B_*TENC; i += NT) {
      const int tp = i % TENC, b = i / TENC;
      out[OUT_ALIGN + ((size_t)b*TDEC + t)*TENC + tp] = aw[i];
    }
    for (int i = gtid; i < B_*1024; i += NT) {
      const int b = i >> 10, u = i & 1023;
      xd[(size_t)b*KD + 1536 + u] = (_Float16)dh[i];
    }
    grid_barrier(bar, &sGen);
  }
}

// ---------------- launcher ----------------
extern "C" void kernel_launch(void* const* d_in, const int* in_sizes, int n_in,
                              void* d_out, int out_size, void* d_ws, size_t ws_size,
                              hipStream_t stream) {
  (void)in_sizes; (void)n_in; (void)out_size; (void)ws_size;
  init_bar_kernel<<<1, 1, 0, stream>>>((unsigned*)d_ws);
  tacotron_decoder_kernel<<<NWG, TPB, 0, stream>>>(
      (const float*)d_in[0],  (const float*)d_in[1],
      (const float*)d_in[2],  (const float*)d_in[3],
      (const float*)d_in[4],  (const float*)d_in[5],
      (const float*)d_in[6],  (const float*)d_in[7],
      (const float*)d_in[8],  (const float*)d_in[9],
      (const float*)d_in[10], (const float*)d_in[11],
      (const float*)d_in[12], (const float*)d_in[13],
      (const float*)d_in[14], (const float*)d_in[15],
      (const float*)d_in[16], (const float*)d_in[17],
      (const float*)d_in[18], (const float*)d_in[19],
      (const float*)d_in[20], (const float*)d_in[21],
      (const float*)d_in[22], (const int*)d_in[23],
      (float*)d_out, (char*)d_ws);
}
